// self_attention_head_61375082660264
// MI455X (gfx1250) — matrix operand order repaired and checked by reading, not compiled, not run on hardware
//
#include <hip/hip_runtime.h>
#include <hip/hip_bf16.h>

typedef __attribute__((ext_vector_type(16))) _Float16 v16h;
typedef __attribute__((ext_vector_type(8)))  float    v8f;

#define B_  4
#define T_  4096
#define C_  512
#define DH_ 64

union U32H2 { unsigned int u; _Float16 h[2]; };

__device__ __forceinline__ v8f wmma_f16(v16h a, v16h b, v8f c) {
  // 8 args: (neg_a, A, neg_b, B, c_mod, C, reuse_a, reuse_b)
  return __builtin_amdgcn_wmma_f32_16x16x32_f16(false, a, false, b, (short)0, c,
                                                false, false);
}

// Async DMA: global -> LDS, 16 bytes per lane, tracked by ASYNCcnt.
// vdst = LDS byte address (32-bit VGPR), vaddr = 64-bit global address.
__device__ __forceinline__ void async_cp16(unsigned lds_byte_addr, const void* gaddr) {
  asm volatile("global_load_async_to_lds_b128 %0, %1, off"
               :: "v"(lds_byte_addr), "v"(gaddr) : "memory");
}
__device__ __forceinline__ unsigned lds_addr32(const void* p) {
  // generic shared pointer: LDS byte offset lives in addr[31:0]
  return (unsigned)(unsigned long long)p;
}

// ---------------------------------------------------------------------------
// Kernel 1: QKV projection.  One wave per 16-token tile; computes the full
// 16x64 output tile of Q, K and V with f32-accumulating f16 WMMAs, then
// stores f16 Q/K row-major and V transposed ([B][64][T]) into workspace.
// ---------------------------------------------------------------------------
__global__ __launch_bounds__(256)
void qkv_proj_f16_wmma(const float* __restrict__ x,
                       const float* __restrict__ Wq, const float* __restrict__ bq,
                       const float* __restrict__ Wk, const float* __restrict__ bk,
                       const float* __restrict__ Wv, const float* __restrict__ bv,
                       _Float16* __restrict__ Qh, _Float16* __restrict__ Kh,
                       _Float16* __restrict__ Vt) {
  const int wave = threadIdx.x >> 5;
  const int lane = threadIdx.x & 31;
  const int half = lane >> 4;
  const int l16  = lane & 15;
  const int tile = blockIdx.x * 8 + wave;   // 16-token tile id
  const int r0   = tile * 16;

  const float* Ws[3] = {Wq, Wk, Wv};
  v8f acc[3][4];
  #pragma unroll
  for (int w = 0; w < 3; ++w)
    #pragma unroll
    for (int nt = 0; nt < 4; ++nt) acc[w][nt] = {};

  const float* xrow = x + (size_t)(r0 + l16) * C_;   // A-matrix: M = l16

  for (int kk = 0; kk < C_; kk += 32) {
    // A fragment: lanes 0-15 hold K 0..15, lanes 16-31 hold K 16..31
    v16h a;
    #pragma unroll
    for (int p = 0; p < 8; ++p) {
      const int kb = kk + (half ? 16 : 0) + 2 * p;
      const float2 f = *(const float2*)(xrow + kb);
      a[2 * p]     = (_Float16)f.x;
      a[2 * p + 1] = (_Float16)f.y;
    }
    #pragma unroll
    for (int w = 0; w < 3; ++w) {
      #pragma unroll
      for (int nt = 0; nt < 4; ++nt) {
        // B fragment: B(k,n) = W[n][k]; lanes 0-15: K 0..15, lanes 16-31: K 16..31
        const float* wrow = Ws[w] + (size_t)(nt * 16 + l16) * C_ + kk + (half ? 16 : 0);
        v16h bf;
        #pragma unroll
        for (int p = 0; p < 8; ++p) {
          const float2 f = *(const float2*)(wrow + 2 * p);
          bf[2 * p]     = (_Float16)f.x;
          bf[2 * p + 1] = (_Float16)f.y;
        }
        acc[w][nt] = wmma_f16(a, bf, acc[w][nt]);
      }
    }
  }

  // Epilogue: bias add (f32), convert to f16.  C layout: N = l16, M = r + 8*half.
  const int bb = r0 / T_;
  const int t0 = r0 - bb * T_;
  const float* Bs[3] = {bq, bk, bv};
  #pragma unroll
  for (int w = 0; w < 3; ++w) {
    #pragma unroll
    for (int nt = 0; nt < 4; ++nt) {
      const int n = nt * 16 + l16;
      const float bias = Bs[w][n];
      #pragma unroll
      for (int r = 0; r < 8; ++r) {
        const int m = r + 8 * half;
        const float v = acc[w][nt][r] + bias;
        if (w == 0)      Qh[(size_t)(r0 + m) * DH_ + n] = (_Float16)v;
        else if (w == 1) Kh[(size_t)(r0 + m) * DH_ + n] = (_Float16)v;
        else             Vt[(size_t)bb * DH_ * T_ + (size_t)n * T_ + (t0 + m)] = (_Float16)v;
      }
    }
  }
}

// ---------------------------------------------------------------------------
// Kernel 2: causal flash attention with double-buffered async K/V staging.
// 4 waves/block own 4 adjacent 16-query tiles.  Per 32-key step the block
// cooperatively DMAs the K tile (32x64 f16) and V^T tile (64x32 f16) into LDS
// with GLOBAL_LOAD_ASYNC_TO_LDS_B128, prefetching step i+1 while step i's
// WMMAs/softmax run.  Pipeline per step:
//   issue copy(i+1) -> buf^1 ; s_wait_asynccnt 4 (copy(i) done, in-order) ;
//   barrier (RAW) ; compute on buf ; barrier (WAR for buf^1).
// ---------------------------------------------------------------------------
__global__ __launch_bounds__(128)
void flash_attn_f16_wmma(const _Float16* __restrict__ Qh,
                         const _Float16* __restrict__ Kh,
                         const _Float16* __restrict__ Vt,
                         float* __restrict__ out) {
  __shared__ __align__(16) _Float16 Kbuf[2][32 * 64];  // [key_local][dim]
  __shared__ __align__(16) _Float16 Vbuf[2][64 * 32];  // [dim][key_local]
  __shared__ __align__(16) _Float16 ldsP[4][16 * 32];  // per-wave P transpose

  const int b    = blockIdx.y;
  const int wave = threadIdx.x >> 5;
  const int lane = threadIdx.x & 31;
  const int half = lane >> 4;
  const int l16  = lane & 15;
  const int q0b  = blockIdx.x * 64;         // block's first query row
  const int q0   = q0b + wave * 16;         // this wave's first query row

  const _Float16* Qb = Qh + (size_t)b * T_ * DH_;
  const _Float16* Kb = Kh + (size_t)b * T_ * DH_;
  const _Float16* Vb = Vt + (size_t)b * DH_ * T_;

  // ---- cooperative async tile copy: 128 threads x 32B each per tile ----
  const int tid = threadIdx.x;
  const int vd  = tid >> 1;                 // V dim row 0..63
  const int vh  = tid & 1;                  // half of the 64B dim row
  auto issue_copy = [&](int kb, int buf) {
    // K tile: rows kb..kb+31, 128B/row, contiguous 4KB block
    const char* gk = (const char*)(Kb + (size_t)kb * DH_) + tid * 32;
    const unsigned lk = lds_addr32((const char*)&Kbuf[buf][0] + tid * 32);
    async_cp16(lk,      gk);
    async_cp16(lk + 16, gk + 16);
    // V^T tile: 64 dim-rows x 32 keys (64B/dim-row), source stride T_ f16
    const char* gv = (const char*)(Vb + (size_t)vd * T_ + kb) + vh * 32;
    const unsigned lv = lds_addr32((const char*)&Vbuf[buf][0] + vd * 64 + vh * 32);
    async_cp16(lv,      gv);
    async_cp16(lv + 16, gv + 16);
  };

  // Q A-fragments (this wave's own rows, straight from global)
  v16h qf[2];
  #pragma unroll
  for (int j = 0; j < 2; ++j) {
    const _Float16* qrow = Qb + (size_t)(q0 + l16) * DH_ + j * 32;
    #pragma unroll
    for (int p = 0; p < 8; ++p) {
      const int kb = (half ? 16 : 0) + 2 * p;
      U32H2 u; u.u = *(const unsigned int*)(qrow + kb);
      qf[j][2 * p] = u.h[0]; qf[j][2 * p + 1] = u.h[1];
    }
  }

  float mrun[8], lrun[8];
  v8f O[4];
  #pragma unroll
  for (int r = 0; r < 8; ++r) { mrun[r] = -__builtin_inff(); lrun[r] = 0.0f; }
  #pragma unroll
  for (int nt = 0; nt < 4; ++nt) O[nt] = {};

  const int nsteps = q0b / 32 + 2;          // keys 0 .. q0b+63
  issue_copy(0, 0);                         // prologue: prefetch step 0

  for (int i = 0; i < nsteps; ++i) {
    const int kb  = i * 32;
    const int cur = i & 1;

    if (i + 1 < nsteps) {
      issue_copy(kb + 32, cur ^ 1);
      asm volatile("s_wait_asynccnt 0x4" ::: "memory");   // copy(i) landed
    } else {
      asm volatile("s_wait_asynccnt 0x0" ::: "memory");
    }
    __syncthreads();                        // RAW: buf[cur] visible block-wide

    if (kb <= q0 + 15) {                    // wave-uniform guard (EXEC stays full)
      // --- scores: two 16x16 tiles from LDS K tile ---
      v8f s[2]; s[0] = {}; s[1] = {};
      #pragma unroll
      for (int t = 0; t < 2; ++t) {
        const _Float16* krow = &Kbuf[cur][(t * 16 + l16) * 64];
        #pragma unroll
        for (int j = 0; j < 2; ++j) {
          v16h kf;
          #pragma unroll
          for (int p = 0; p < 8; ++p) {
            U32H2 u; u.u = *(const unsigned int*)(krow + j * 32 + (half ? 16 : 0) + 2 * p);
            kf[2 * p] = u.h[0]; kf[2 * p + 1] = u.h[1];
          }
          s[t] = wmma_f16(qf[j], kf, s[t]);
        }
      }

      // --- causal mask + online softmax (rows replicated across 16 lanes) ---
      float corr[8];
      #pragma unroll
      for (int r = 0; r < 8; ++r) {
        const int q = q0 + r + 8 * half;
        float s0 = s[0][r];
        float s1 = s[1][r];
        if (kb +      l16 > q) s0 = -__builtin_inff();
        if (kb + 16 + l16 > q) s1 = -__builtin_inff();
        float tm = fmaxf(s0, s1);
        #pragma unroll
        for (int off = 1; off < 16; off <<= 1)
          tm = fmaxf(tm, __shfl_xor(tm, off));
        const float mnew = fmaxf(mrun[r], tm);
        corr[r] = __expf(mrun[r] - mnew);
        mrun[r] = mnew;
        const float p0 = __expf(s0 - mnew);
        const float p1 = __expf(s1 - mnew);
        const int m = r + 8 * half;
        ldsP[wave][m * 32 +      l16] = (_Float16)p0;
        ldsP[wave][m * 32 + 16 + l16] = (_Float16)p1;
        float rs = p0 + p1;
        #pragma unroll
        for (int off = 1; off < 16; off <<= 1)
          rs += __shfl_xor(rs, off);
        lrun[r] = lrun[r] * corr[r] + rs;
      }
      #pragma unroll
      for (int nt = 0; nt < 4; ++nt)
        #pragma unroll
        for (int r = 0; r < 8; ++r) O[nt][r] *= corr[r];

      asm volatile("s_wait_dscnt 0x0" ::: "memory");  // P stores -> P reloads

      // P as a 16x32 f16 A-fragment (cross-lane transpose via LDS)
      v16h pf;
      #pragma unroll
      for (int p = 0; p < 8; ++p) {
        const int off = (half ? 16 : 0) + 2 * p;
        U32H2 u; u.u = *(const unsigned int*)&ldsP[wave][l16 * 32 + off];
        pf[2 * p] = u.h[0]; pf[2 * p + 1] = u.h[1];
      }

      // --- O += P * V from LDS V^T tile ---
      #pragma unroll
      for (int nt = 0; nt < 4; ++nt) {
        const _Float16* vrow = &Vbuf[cur][(nt * 16 + l16) * 32 + (half ? 16 : 0)];
        v16h vf;
        #pragma unroll
        for (int p = 0; p < 8; ++p) {
          U32H2 u; u.u = *(const unsigned int*)(vrow + 2 * p);
          vf[2 * p] = u.h[0]; vf[2 * p + 1] = u.h[1];
        }
        O[nt] = wmma_f16(pf, vf, O[nt]);
      }
    }

    __syncthreads();   // WAR: everyone done reading buf[cur] before it's re-filled
  }

  // Epilogue: out = O * sqrt(d_h) / l = O * 8 / l   (fp32)
  float* orow = out + (size_t)b * T_ * DH_;
  #pragma unroll
  for (int nt = 0; nt < 4; ++nt) {
    #pragma unroll
    for (int r = 0; r < 8; ++r) {
      const int m = r + 8 * half;
      orow[(size_t)(q0 + m) * DH_ + nt * 16 + l16] = O[nt][r] * (8.0f / lrun[r]);
    }
  }
}

// ---------------------------------------------------------------------------
extern "C" void kernel_launch(void* const* d_in, const int* in_sizes, int n_in,
                              void* d_out, int out_size, void* d_ws, size_t ws_size,
                              hipStream_t stream) {
  const float* x  = (const float*)d_in[0];
  const float* Wq = (const float*)d_in[1];
  const float* bq = (const float*)d_in[2];
  const float* Wk = (const float*)d_in[3];
  const float* bk = (const float*)d_in[4];
  const float* Wv = (const float*)d_in[5];
  const float* bv = (const float*)d_in[6];
  float* out = (float*)d_out;

  // Workspace: Qh, Kh row-major f16 [B*T,64]; V transposed f16 [B][64][T].
  _Float16* Qh = (_Float16*)d_ws;
  _Float16* Kh = Qh + (size_t)B_ * T_ * DH_;
  _Float16* Vt = Kh + (size_t)B_ * T_ * DH_;

  qkv_proj_f16_wmma<<<dim3((B_ * T_) / (16 * 8)), 256, 0, stream>>>(
      x, Wq, bq, Wk, bk, Wv, bv, Qh, Kh, Vt);

  flash_attn_f16_wmma<<<dim3(T_ / 64, B_), 128, 0, stream>>>(Qh, Kh, Vt, out);
}
